// Transducer_35184372088842
// MI455X (gfx1250) — matrix operand order repaired and checked by reading, not compiled, not run on hardware
//
#include <hip/hip_runtime.h>
#include <math.h>

#define Bc     4
#define Tt     256
#define Uu     64
#define Hh     512
#define INNERD 512
#define VOCABD 1024

typedef __attribute__((ext_vector_type(16))) __bf16 v16bf;
typedef __attribute__((ext_vector_type(8)))  float  v8f;

union Frag16 {
    v16bf v;
    uint4 q[2];
};

__device__ __forceinline__ unsigned short f2bf(float f) {
    unsigned u = __builtin_bit_cast(unsigned, f);
    unsigned r = (u + 0x7FFFu + ((u >> 16) & 1u)) >> 16;   // round-to-nearest-even
    return (unsigned short)r;
}

// ---------------------------------------------------------------------------
// Kernel 0: convert W2 [512,1024] f32 -> bf16, pre-swizzled into WMMA B-fragment
// layout.  Tile (kt,nt) holds 32(K) x 16(N); lane l, element e maps to
//   k = kt*32 + 16*(l>>4) + e,   n = nt*16 + (l&15)
// so the GEMM kernel reads each fragment as 2 coalesced uint4 per lane.
// ---------------------------------------------------------------------------
__global__ __launch_bounds__(256) void pack_w2_kernel(
    const float* __restrict__ W2, unsigned short* __restrict__ w2p)
{
    int tid = blockIdx.x * 256 + threadIdx.x;   // 0 .. 32767
    int l   = tid & 31;
    int nt  = (tid >> 5) & 63;
    int kt  = tid >> 11;
    int kb  = kt * 32 + ((l >> 4) * 16);
    int n   = nt * 16 + (l & 15);

    union { unsigned short u[16]; uint4 q[2]; } pk;
#pragma unroll
    for (int e = 0; e < 16; ++e)
        pk.u[e] = f2bf(W2[(size_t)(kb + e) * VOCABD + n]);

    uint4* p  = (uint4*)w2p;
    int   idx = (kt * 64 + nt) * 64 + l * 2;
    p[idx]     = pk.q[0];
    p[idx + 1] = pk.q[1];
}

// ---------------------------------------------------------------------------
// Kernel 1: f32 projections  encP[b,t,:] = enc @ W1[:H],  decP[b,u,:] = dec @ W1[H:]
// One block per output row; source row staged in LDS, W1 reads coalesced.
// ---------------------------------------------------------------------------
__global__ __launch_bounds__(256) void proj_kernel(
    const float* __restrict__ enc, const float* __restrict__ dec,
    const float* __restrict__ W1,
    float* __restrict__ encP, float* __restrict__ decP)
{
    __shared__ float srow[Hh];
    const int r   = blockIdx.x;
    const int tid = threadIdx.x;

    const float* src;
    const float* wb;
    float*       dst;
    if (r < Bc * Tt) {
        src = enc + (size_t)r * Hh;
        wb  = W1;
        dst = encP + (size_t)r * INNERD;
    } else {
        int rd = r - Bc * Tt;
        src = dec + (size_t)rd * Hh;
        wb  = W1 + (size_t)Hh * INNERD;
        dst = decP + (size_t)rd * INNERD;
    }
    srow[tid]       = src[tid];
    srow[tid + 256] = src[tid + 256];
    __syncthreads();

    const int i0 = tid * 2;
    float a0 = 0.f, a1 = 0.f;
#pragma unroll 4
    for (int k = 0; k < Hh; ++k) {
        float  s  = srow[k];
        float2 wv = *(const float2*)(wb + (size_t)k * INNERD + i0);
        a0 = fmaf(s, wv.x, a0);
        a1 = fmaf(s, wv.y, a1);
    }
    float2 o; o.x = a0; o.y = a1;
    *(float2*)(dst + i0) = o;
}

// ---------------------------------------------------------------------------
// Kernel 2: fused  tanh(encP+decP+b1) @ W2 + b2  -> log_softmax -> out
// Block = (b, t, u-block of 16): M=16, N=1024, K=512 GEMM via
// v_wmma_f32_16x16x32_bf16.  8 waves x 8 N-tiles; softmax entirely in regs +
// shfl_xor (wave32) + tiny LDS cross-wave reduce.  Single 64KB global write.
// ---------------------------------------------------------------------------
__global__ __launch_bounds__(256) void joint_fused_kernel(
    const float* __restrict__ encP, const float* __restrict__ decP,
    const float* __restrict__ b1,   const unsigned short* __restrict__ w2p,
    const float* __restrict__ b2,   float* __restrict__ out)
{
    __shared__ __align__(16) unsigned short Alds[16 * 520];  // 16 rows x 512 bf16, padded
    __shared__ float redMax[8][16];
    __shared__ float redSum[8][16];

    const int tid = threadIdx.x;
    const int bid = blockIdx.x;
    const int ub  = bid & 3;            // U/16 = 4
    const int t   = (bid >> 2) & 255;
    const int b   = bid >> 10;
    const int u0  = ub * 16;

    // ---- Phase 1: build A panel: hidden[m,k] = tanh(encP+decP+b1) as bf16 ----
    {
        const int row = tid >> 4;
        const int sub = tid & 15;
        const float* er = encP + (size_t)(b * Tt + t) * INNERD;
        const float* dr = decP + (size_t)(b * Uu + u0 + row) * INNERD;
#pragma unroll
        for (int j = 0; j < 32; j += 4) {
            int    k  = sub * 32 + j;
            float4 e4 = *(const float4*)(er + k);
            float4 d4 = *(const float4*)(dr + k);
            float4 c4 = *(const float4*)(b1 + k);
            unsigned short h0 = f2bf(tanhf(e4.x + d4.x + c4.x));
            unsigned short h1 = f2bf(tanhf(e4.y + d4.y + c4.y));
            unsigned short h2 = f2bf(tanhf(e4.z + d4.z + c4.z));
            unsigned short h3 = f2bf(tanhf(e4.w + d4.w + c4.w));
            uint2 pk;
            pk.x = (unsigned)h0 | ((unsigned)h1 << 16);
            pk.y = (unsigned)h2 | ((unsigned)h3 << 16);
            *(uint2*)&Alds[row * 520 + k] = pk;
        }
    }
    __syncthreads();

    // ---- Phase 2: GEMM (M=16, N=128 per wave, K=512) ----
    const int lane = tid & 31;
    const int w    = tid >> 5;
    const int row  = lane & 15;
    const int half = lane >> 4;
    const int klo  = half * 16;    // contiguous k order (as in w2p): upper half starts at K=16

    v8f acc[8];
#pragma unroll
    for (int s = 0; s < 8; ++s)
#pragma unroll
        for (int e = 0; e < 8; ++e) acc[s][e] = 0.f;

    const uint4* gB = (const uint4*)w2p;
    const int    bl = lane * 2;

    for (int kt = 0; kt < 16; ++kt) {
        Frag16 a;
        const unsigned short* ap = &Alds[row * 520 + kt * 32 + klo];
        a.q[0] = *(const uint4*)ap;          // K = klo .. klo+7
        a.q[1] = *(const uint4*)(ap + 8);    // K = klo+8 .. klo+15
#pragma unroll
        for (int s = 0; s < 8; ++s) {
            int    nt = w * 8 + s;
            Frag16 bf;
            const uint4* bp = gB + ((kt * 64 + nt) * 64 + bl);
            bf.q[0] = bp[0];
            bf.q[1] = bp[1];
            acc[s] = __builtin_amdgcn_wmma_f32_16x16x32_bf16(
                false, a.v, false, bf.v, (short)0, acc[s], false, false);
        }
    }

    // ---- Phase 3: + b2, then log_softmax reductions ----
    const int col = lane & 15;
#pragma unroll
    for (int s = 0; s < 8; ++s) {
        float bb = b2[(w * 8 + s) * 16 + col];
#pragma unroll
        for (int e = 0; e < 8; ++e) acc[s][e] += bb;
    }

    // per-lane row maxima (8 rows per lane), reduce across the 16-lane half
    float fmx[8];
#pragma unroll
    for (int e = 0; e < 8; ++e) {
        float m = acc[0][e];
#pragma unroll
        for (int s = 1; s < 8; ++s) m = fmaxf(m, acc[s][e]);
#pragma unroll
        for (int off = 8; off >= 1; off >>= 1)
            m = fmaxf(m, __shfl_xor(m, off));
        fmx[e] = m;
    }
    if (col == 0) {
#pragma unroll
        for (int e = 0; e < 8; ++e) redMax[w][half * 8 + e] = fmx[e];
    }
    __syncthreads();
#pragma unroll
    for (int e = 0; e < 8; ++e) {
        int   m  = half * 8 + e;
        float mm = redMax[0][m];
#pragma unroll
        for (int ww = 1; ww < 8; ++ww) mm = fmaxf(mm, redMax[ww][m]);
        fmx[e] = mm;
    }

    float lse[8];
#pragma unroll
    for (int e = 0; e < 8; ++e) {
        float sum = 0.f;
#pragma unroll
        for (int s = 0; s < 8; ++s) sum += expf(acc[s][e] - fmx[e]);
#pragma unroll
        for (int off = 8; off >= 1; off >>= 1)
            sum += __shfl_xor(sum, off);
        lse[e] = sum;
    }
    if (col == 0) {
#pragma unroll
        for (int e = 0; e < 8; ++e) redSum[w][half * 8 + e] = lse[e];
    }
    __syncthreads();
#pragma unroll
    for (int e = 0; e < 8; ++e) {
        int   m  = half * 8 + e;
        float ss = 0.f;
#pragma unroll
        for (int ww = 0; ww < 8; ++ww) ss += redSum[ww][m];
        lse[e] = logf(ss);
    }

    // ---- Phase 4: write out = logits - max - log(sum) ----
    float* orow = out + (size_t)((b * Tt + t) * Uu + u0) * VOCABD;
#pragma unroll
    for (int s = 0; s < 8; ++s) {
        int n = (w * 8 + s) * 16 + col;
#pragma unroll
        for (int e = 0; e < 8; ++e) {
            int m = half * 8 + e;
            orow[(size_t)m * VOCABD + n] = acc[s][e] - fmx[e] - lse[e];
        }
    }
}

// ---------------------------------------------------------------------------
extern "C" void kernel_launch(void* const* d_in, const int* in_sizes, int n_in,
                              void* d_out, int out_size, void* d_ws, size_t ws_size,
                              hipStream_t stream) {
    const float* enc = (const float*)d_in[0];
    const float* dec = (const float*)d_in[1];
    const float* W1  = (const float*)d_in[2];
    const float* b1  = (const float*)d_in[3];
    const float* W2  = (const float*)d_in[4];
    const float* b2  = (const float*)d_in[5];
    float*       out = (float*)d_out;

    // workspace layout
    unsigned short* w2p  = (unsigned short*)d_ws;                       // 1 MiB
    float*          encP = (float*)((char*)d_ws + (1u << 20));          // 2 MiB
    float*          decP = (float*)((char*)d_ws + (1u << 20) + (2u << 20)); // 0.5 MiB

    pack_w2_kernel<<<128, 256, 0, stream>>>(W2, w2p);
    proj_kernel<<<Bc * Tt + Bc * Uu, 256, 0, stream>>>(enc, dec, W1, encP, decP);
    joint_fused_kernel<<<Bc * Tt * (Uu / 16), 256, 0, stream>>>(encP, decP, b1, w2p, b2, out);
}
